// GroupedQueryAttention_56264071577706
// MI455X (gfx1250) — matrix operand order repaired and checked by reading, not compiled, not run on hardware
//
#include <hip/hip_runtime.h>
#include <hip/hip_bf16.h>

typedef __bf16 bf16;
typedef __attribute__((ext_vector_type(8)))  __bf16 v8bf;
typedef __attribute__((ext_vector_type(16))) __bf16 v16bf;
typedef __attribute__((ext_vector_type(8)))  float  v8f;
typedef __attribute__((ext_vector_type(4)))  unsigned int uint32x4;
typedef __attribute__((ext_vector_type(8)))  int int32x8;
typedef __attribute__((ext_vector_type(4)))  int int32x4;

// ---------------- problem constants ----------------
constexpr int B_SZ   = 4;
constexpr int T_SEQ  = 2048;
constexpr int NEMB   = 1536;
constexpr int HKV    = 8;
constexpr int DQK    = 192;   // per-group q/k dim after tiling
constexpr int DV     = 64;    // effective v dim (output = 3 replicas)
constexpr int GEMM_M = B_SZ * T_SEQ;      // 8192
constexpr int GEMM_N = NEMB + 2 * 512;    // 2560  (q 1536 | k 512 | v 512)
constexpr int GEMM_K = NEMB;              // 1536
constexpr float QSCALE = 0.07216878364870323f;  // 1/sqrt(192)

// ---------------- WMMA helpers (layouts per CDNA5 ISA) ----------------
__device__ __forceinline__ v8f wmma_bf16(v16bf a, v16bf b, v8f c) {
  return __builtin_amdgcn_wmma_f32_16x16x32_bf16(
      false, a, false, b, (short)0, c, false, false);
}

// A fragment 16x32 bf16 from row-major (M x K) LDS tile, stride ldk elems.
__device__ __forceinline__ v16bf load_frag_a(const bf16* base, int ldk) {
  int l   = threadIdx.x & 31;
  int row = l & 15;
  int kb  = (l >> 4) * 16;
  const bf16* p = base + row * ldk + kb;
  v8bf lo = *(const v8bf*)p;
  v8bf hi = *(const v8bf*)(p + 8);
  return __builtin_shufflevector(lo, hi, 0,1,2,3,4,5,6,7,8,9,10,11,12,13,14,15);
}

// B fragment 32x16 bf16 from N-major LDS tile: element (k,n) at base[n*ldk + k].
__device__ __forceinline__ v16bf load_frag_b(const bf16* base, int ldk) {
  int l  = threadIdx.x & 31;
  int n  = l & 15;
  int kb = (l >> 4) * 16;
  const bf16* p = base + n * ldk + kb;
  v8bf lo = *(const v8bf*)p;
  v8bf hi = *(const v8bf*)(p + 8);
  return __builtin_shufflevector(lo, hi, 0,1,2,3,4,5,6,7,8,9,10,11,12,13,14,15);
}

// B fragment from TDM-padded tile: rows of 192 elems, 16B pad per 128B
// -> row stride 216 elems, each 64-elem block strides 72 elems.
__device__ __forceinline__ v16bf load_frag_b_pad(const bf16* kbase, int t0, int k0) {
  int l  = threadIdx.x & 31;
  int n  = t0 + (l & 15);
  int kk = k0 + (l >> 4) * 16;
  const bf16* p = kbase + n * 216 + (kk >> 6) * 72 + (kk & 63);
  v8bf lo = *(const v8bf*)p;
  v8bf hi = *(const v8bf*)(p + 8);
  return __builtin_shufflevector(lo, hi, 0,1,2,3,4,5,6,7,8,9,10,11,12,13,14,15);
}

// Async 16-byte global -> LDS copy (GLOBAL_LOAD_ASYNC_TO_LDS_B128, ASYNCcnt)
__device__ __forceinline__ void async_cp16(bf16* dst_lds, const bf16* src_glb) {
  asm volatile("global_load_async_to_lds_b128 %0, %1, off"
               :: "v"((unsigned int)(unsigned long long)dst_lds), "v"(src_glb)
               : "memory");
}
__device__ __forceinline__ void async_wait0() {
  asm volatile("s_wait_asynccnt 0" ::: "memory");
}

// Tensor Data Mover: 2D tile load global->LDS with LDS padding
// (16B pad after every 128B; CDNA5 ISA D# groups).
// tile0 = contiguous elems per row, tile1 = rows, stride0 = global row stride (elems)
__device__ __forceinline__ void tdm_load_2d(bf16* lds, const bf16* gsrc,
                                            unsigned tile0, unsigned tile1,
                                            unsigned long long stride0) {
  unsigned long long ga = (unsigned long long)gsrc;
  uint32x4 g0;
  g0[0] = 1u;                                         // count=1, user mode
  g0[1] = (unsigned int)(unsigned long long)lds;      // lds_addr (bytes)
  g0[2] = (unsigned int)ga;                           // global_addr[31:0]
  g0[3] = (unsigned int)(ga >> 32) | (2u << 30);      // global_addr[56:32] | type=2
  int32x8 g1;
  g1[0] = (1 << 16) | (1 << 20) | (4 << 22) | (3 << 25);
      // data_size=1 (2B) | pad_enable | pad_interval=32dw | pad_amount=4dw
  g1[1] = (int)((tile0 & 0xffffu) << 16);             // tensor_dim0 = tile0
  g1[2] = (int)((tile1 & 0xffffu) << 16);             // dim0 hi=0 | tensor_dim1 lo
  g1[3] = (int)((tile0 & 0xffffu) << 16);             // dim1 hi=0 | tile_dim0
  g1[4] = (int)(tile1 & 0xffffu);                     // tile_dim1 | tile_dim2=0
  g1[5] = (int)(unsigned int)stride0;                 // tensor_dim0_stride lo32
  g1[6] = (int)((stride0 >> 32) & 0xffffu);           // stride hi16 | dim1_stride lo=0
  g1[7] = 0;
  int32x4 z4 = {0, 0, 0, 0};
  int32x8 z8 = {0, 0, 0, 0, 0, 0, 0, 0};
  __builtin_amdgcn_tensor_load_to_lds(g0, g1, z4, z4, z8, 0);
}

// ---------------- kernel 1: x -> bf16 ----------------
__global__ void cvt_x(const float* __restrict__ x, bf16* __restrict__ xb, int n) {
  int i = blockIdx.x * 256 + threadIdx.x;
  if (i < n) xb[i] = (bf16)x[i];
}

// ---------------- kernel 2: pack+transpose [Wq|Wk|Wv] -> bf16 N-major (2560 x 1536) ----------------
__global__ __launch_bounds__(256) void build_wT(const float* __restrict__ Wq,
                                                const float* __restrict__ Wk,
                                                const float* __restrict__ Wv,
                                                bf16* __restrict__ WT) {
  __shared__ float tile[32][33];
  const int n0 = blockIdx.x * 32;
  const int k0 = blockIdx.y * 32;
  const int tx = threadIdx.x & 31;
  const int ty = threadIdx.x >> 5;
#pragma unroll
  for (int s = 0; s < 4; ++s) {
    int k = k0 + ty + 8 * s;
    int n = n0 + tx;
    float v;
    if (n < 1536)      v = Wq[(size_t)k * 1536 + n];
    else if (n < 2048) v = Wk[(size_t)k * 512 + (n - 1536)];
    else               v = Wv[(size_t)k * 512 + (n - 2048)];
    tile[ty + 8 * s][tx] = v;
  }
  __syncthreads();
#pragma unroll
  for (int s = 0; s < 4; ++s) {
    int n = n0 + ty + 8 * s;
    int k = k0 + tx;
    WT[(size_t)n * GEMM_K + k] = (bf16)tile[tx][ty + 8 * s];
  }
}

// ---------------- kernel 3: RoPE cos/sin table (t x 96 pairs) ----------------
__global__ void build_tab(float2* __restrict__ tab) {
  int t = blockIdx.x, j = threadIdx.x;
  float inv = __expf(-(2.0f * (float)j / 192.0f) * 9.210340371976184f);
  float s, c;
  sincosf((float)t * inv, &s, &c);
  tab[t * 96 + j] = make_float2(c, s);
}

// ---------------- kernel 4: fused QKV GEMM (TDM-staged) + bias + RoPE scatter ----------------
constexpr int GLDK = 72;  // 64 + 8: matches TDM pad layout (144B row stride)

__global__ __launch_bounds__(256) void qkv_rope_gemm(const bf16* __restrict__ X,
                                                     const bf16* __restrict__ WT,
                                                     const float* __restrict__ bq,
                                                     const float* __restrict__ bk,
                                                     const float* __restrict__ bv,
                                                     const float2* __restrict__ tab,
                                                     bf16* __restrict__ Qr,
                                                     bf16* __restrict__ Kr,
                                                     bf16* __restrict__ Vr) {
  alignas(16) __shared__ bf16 As[128 * GLDK];   // M-major, TDM-filled
  alignas(16) __shared__ bf16 Bs[128 * GLDK];   // N-major, TDM-filled
  const int bm = blockIdx.x * 128;
  const int bn = blockIdx.y * 128;
  const int tid  = threadIdx.x;
  const int wave = tid >> 5;
  const int lane = tid & 31;
  const int hl   = lane >> 4;
  const int ln   = lane & 15;
  const int wm = (wave & 3) * 32;
  const int wn = (wave >> 2) * 64;

  v8f acc[2][4] = {};

  for (int k0 = 0; k0 < GEMM_K; k0 += 64) {
    __syncthreads();   // all waves done reading previous stage
    if (wave == 0) {
      tdm_load_2d(As, &X[(size_t)bm * GEMM_K + k0], 64, 128, GEMM_K);
      tdm_load_2d(Bs, &WT[(size_t)bn * GEMM_K + k0], 64, 128, GEMM_K);
      __builtin_amdgcn_s_wait_tensorcnt(0);
    }
    __syncthreads();

#pragma unroll
    for (int kc = 0; kc < 2; ++kc) {
      v16bf a0 = load_frag_a(&As[(wm)      * GLDK + kc * 32], GLDK);
      v16bf a1 = load_frag_a(&As[(wm + 16) * GLDK + kc * 32], GLDK);
#pragma unroll
      for (int j = 0; j < 4; ++j) {
        v16bf b = load_frag_b(&Bs[(wn + 16 * j) * GLDK + kc * 32], GLDK);
        acc[0][j] = wmma_bf16(a0, b, acc[0][j]);
        acc[1][j] = wmma_bf16(a1, b, acc[1][j]);
      }
    }
  }

  // ---- fused epilogue: bias + RoPE + scatter. Region is uniform per block. ----
  if (bn < 1536) {
#pragma unroll
    for (int j = 0; j < 4; ++j) {
      int n = bn + wn + 16 * j + ln;
      int g = n / 192, d = n % 192, jj = d >> 1;
      float bias = bq[n];
#pragma unroll
      for (int mi = 0; mi < 2; ++mi)
#pragma unroll
        for (int r = 0; r < 8; ++r) {
          int m = bm + wm + 16 * mi + r + 8 * hl;
          int b = m >> 11, t = m & 2047;
          float v = acc[mi][j][r] + bias;
          float w = __shfl_xor(v, 1);                  // rope pair partner
          float2 cs = tab[t * 96 + jj];
          float o = (ln & 1) ? (w * cs.y + v * cs.x)   // odd: yr*sin + yi*cos
                             : (v * cs.x - w * cs.y);  // even: yr*cos - yi*sin
          Qr[((size_t)(b * HKV + g) * T_SEQ + t) * DQK + d] = (bf16)(o * QSCALE);
        }
    }
  } else if (bn < 2048) {
#pragma unroll
    for (int j = 0; j < 4; ++j) {
      int n = bn + wn + 16 * j + ln;
      int kk = n - 1536;
      int g = kk / 64, dd = kk % 64, jj0 = dd >> 1;
      float bias = bk[kk];
#pragma unroll
      for (int mi = 0; mi < 2; ++mi)
#pragma unroll
        for (int r = 0; r < 8; ++r) {
          int m = bm + wm + 16 * mi + r + 8 * hl;
          int b = m >> 11, t = m & 2047;
          float v = acc[mi][j][r] + bias;
          float w = __shfl_xor(v, 1);
          size_t base = ((size_t)(b * HKV + g) * T_SEQ + t) * DQK + dd;
#pragma unroll
          for (int rep = 0; rep < 3; ++rep) {
            float2 cs = tab[t * 96 + jj0 + 32 * rep];
            float o = (ln & 1) ? (w * cs.y + v * cs.x)
                               : (v * cs.x - w * cs.y);
            Kr[base + 64 * rep] = (bf16)o;
          }
        }
    }
  } else {
#pragma unroll
    for (int j = 0; j < 4; ++j) {
      int n = bn + wn + 16 * j + ln;
      int vv = n - 2048;
      int g = vv / 64, dd = vv % 64;
      float bias = bv[vv];
#pragma unroll
      for (int mi = 0; mi < 2; ++mi)
#pragma unroll
        for (int r = 0; r < 8; ++r) {
          int m = bm + wm + 16 * mi + r + 8 * hl;
          int b = m >> 11, t = m & 2047;
          Vr[((size_t)(b * HKV + g) * T_SEQ + t) * DV + dd] =
              (bf16)(acc[mi][j][r] + bias);
        }
    }
  }
}

// ---------------- kernel 5: causal flash attention ----------------
// Q: async-LDS staged; K: TDM-staged (padded layout); V: normal (transpose)
constexpr int QTILE = 128;
constexpr int KTILE = 64;
constexpr int LDQ = 200;   // 192+8 pad
constexpr int KROW = 216;  // TDM layout: 192 data + 3x8 pad elems per row
constexpr int LDV = 72;
constexpr int LDP = 72;

__global__ __launch_bounds__(256) void attn(const bf16* __restrict__ Qr,
                                            const bf16* __restrict__ Kr,
                                            const bf16* __restrict__ Vr,
                                            float* __restrict__ out) {
  alignas(16) __shared__ bf16 Qs[QTILE * LDQ];    // 50 KB
  alignas(16) __shared__ bf16 Ks[KTILE * KROW];   // 27 KB (TDM-filled)
  alignas(16) __shared__ bf16 Vts[DV * LDV];      //  9 KB (V transposed [d][t])
  alignas(16) __shared__ bf16 Ps[QTILE * LDP];    // 18 KB

  const int qt  = blockIdx.x;
  const int bg  = blockIdx.y;
  const int q0  = qt * QTILE;
  const int tid = threadIdx.x;
  const int wave = tid >> 5;
  const int lane = tid & 31;
  const int hl = lane >> 4, ln = lane & 15;

  const bf16* Qg = Qr + ((size_t)bg * T_SEQ + q0) * DQK;
  const bf16* Kg = Kr + (size_t)bg * T_SEQ * DQK;
  const bf16* Vg = Vr + (size_t)bg * T_SEQ * DV;

  // async-stage the Q tile (128 x 192)
  for (int i = tid; i < QTILE * (DQK / 8); i += 256) {
    int r = i / (DQK / 8), c = (i % (DQK / 8)) * 8;
    async_cp16(&Qs[r * LDQ + c], &Qg[(size_t)r * DQK + c]);
  }
  async_wait0();

  v8f o[4] = {};
  v8f mrow, denom;
#pragma unroll
  for (int r = 0; r < 8; ++r) { mrow[r] = -3.0e38f; denom[r] = 0.0f; }

  const int nkv = (q0 + QTILE) / KTILE;

  for (int kt = 0; kt < nkv; ++kt) {
    // TDM-stage K tile (64 rows x 192, padded to KROW stride in LDS)
    if (wave == 0)
      tdm_load_2d(Ks, &Kg[(size_t)(kt * KTILE) * DQK], DQK, KTILE, DQK);
    // V tile transposed into LDS (element transpose -> normal path)
    for (int i = tid; i < KTILE * DV; i += 256) {
      int r = i / DV, c = i % DV;
      Vts[c * LDV + r] = Vg[((size_t)(kt * KTILE + r)) * DV + c];
    }
    if (wave == 0) __builtin_amdgcn_s_wait_tensorcnt(0);
    __syncthreads();

    // S = Q(16x192) . K^T : 24 WMMAs per wave
    v8f s[4] = {};
    const bf16* qbase = &Qs[(wave * 16) * LDQ];
#pragma unroll
    for (int kc = 0; kc < 6; ++kc) {
      v16bf a = load_frag_a(qbase + kc * 32, LDQ);
#pragma unroll
      for (int j = 0; j < 4; ++j) {
        v16bf bb = load_frag_b_pad(Ks, 16 * j, kc * 32);
        s[j] = wmma_bf16(a, bb, s[j]);
      }
    }

    // causal mask + online softmax (row = VGPR idx + 8*hl, col = 16j + ln)
    v8f mnew = mrow;
#pragma unroll
    for (int j = 0; j < 4; ++j) {
      int col = kt * KTILE + 16 * j + ln;
#pragma unroll
      for (int r = 0; r < 8; ++r) {
        int row = q0 + wave * 16 + r + 8 * hl;
        float sv = (col > row) ? -3.0e38f : s[j][r];
        s[j][r] = sv;
        mnew[r] = fmaxf(mnew[r], sv);
      }
    }
#pragma unroll
    for (int r = 0; r < 8; ++r) {
      float v = mnew[r];
      v = fmaxf(v, __shfl_xor(v, 1));
      v = fmaxf(v, __shfl_xor(v, 2));
      v = fmaxf(v, __shfl_xor(v, 4));
      v = fmaxf(v, __shfl_xor(v, 8));
      mnew[r] = v;
    }

    v8f alpha, rsum = {};
#pragma unroll
    for (int r = 0; r < 8; ++r) alpha[r] = __expf(mrow[r] - mnew[r]);
    mrow = mnew;
#pragma unroll
    for (int j = 0; j < 4; ++j)
#pragma unroll
      for (int r = 0; r < 8; ++r) {
        float p = __expf(s[j][r] - mnew[r]);
        s[j][r] = p;
        rsum[r] += p;
      }
#pragma unroll
    for (int r = 0; r < 8; ++r) {
      float v = rsum[r];
      v += __shfl_xor(v, 1);
      v += __shfl_xor(v, 2);
      v += __shfl_xor(v, 4);
      v += __shfl_xor(v, 8);
      denom[r] = denom[r] * alpha[r] + v;
    }
#pragma unroll
    for (int j = 0; j < 4; ++j)
#pragma unroll
      for (int r = 0; r < 8; ++r) o[j][r] *= alpha[r];

    // stage P (bf16) to re-fragment as A
#pragma unroll
    for (int j = 0; j < 4; ++j)
#pragma unroll
      for (int r = 0; r < 8; ++r)
        Ps[(wave * 16 + r + 8 * hl) * LDP + 16 * j + ln] = (bf16)s[j][r];
    __syncthreads();

    // O += P(16x64) . V(64x64) : 8 WMMAs per wave
    v16bf pa0 = load_frag_a(&Ps[(wave * 16) * LDP + 0],  LDP);
    v16bf pa1 = load_frag_a(&Ps[(wave * 16) * LDP + 32], LDP);
#pragma unroll
    for (int j = 0; j < 4; ++j) {
      o[j] = wmma_bf16(pa0, load_frag_b(&Vts[(16 * j) * LDV + 0],  LDV), o[j]);
      o[j] = wmma_bf16(pa1, load_frag_b(&Vts[(16 * j) * LDV + 32], LDV), o[j]);
    }
    __syncthreads();   // protect Ks/Vts/Ps before next stage
  }

  // finalize + write 3 replicas (tiled-V makes the 3 output segments identical)
  const int b = bg >> 3, g = bg & 7;
#pragma unroll
  for (int r = 0; r < 8; ++r) denom[r] = 1.0f / denom[r];
#pragma unroll
  for (int j = 0; j < 4; ++j)
#pragma unroll
    for (int r = 0; r < 8; ++r) {
      int t = q0 + wave * 16 + r + 8 * hl;
      float v = o[j][r] * denom[r];
      float* po = out + ((size_t)(b * T_SEQ + t) * NEMB) + g * DQK + 16 * j + ln;
      po[0] = v; po[64] = v; po[128] = v;
    }
}

// ---------------- host ----------------
extern "C" void kernel_launch(void* const* d_in, const int* in_sizes, int n_in,
                              void* d_out, int out_size, void* d_ws, size_t ws_size,
                              hipStream_t stream) {
  const float* x  = (const float*)d_in[0];
  const float* Wq = (const float*)d_in[1];
  const float* bq = (const float*)d_in[2];
  const float* Wk = (const float*)d_in[3];
  const float* bk = (const float*)d_in[4];
  const float* Wv = (const float*)d_in[5];
  const float* bv = (const float*)d_in[6];
  float* out = (float*)d_out;

  char* ws = (char*)d_ws;
  bf16*   Xb  = (bf16*)(ws);                 // 8192*1536*2    = 25,165,824
  bf16*   WT  = (bf16*)(ws + 25165824);      // 2560*1536*2    =  7,864,320
  float2* tab = (float2*)(ws + 33030144);    // 2048*96*8      =  1,572,864
  bf16*   Qr  = (bf16*)(ws + 34603008);      // 4*8*2048*192*2 = 25,165,824
  bf16*   Kr  = (bf16*)(ws + 59768832);      // same           = 25,165,824
  bf16*   Vr  = (bf16*)(ws + 84934656);      // 4*8*2048*64*2  =  8,388,608
                                             // total ~ 93.3 MB

  const int nx = GEMM_M * GEMM_K;
  cvt_x<<<(nx + 255) / 256, 256, 0, stream>>>(x, Xb, nx);

  build_wT<<<dim3(GEMM_N / 32, GEMM_K / 32), 256, 0, stream>>>(Wq, Wk, Wv, WT);

  build_tab<<<T_SEQ, 96, 0, stream>>>(tab);

  qkv_rope_gemm<<<dim3(GEMM_M / 128, GEMM_N / 128), 256, 0, stream>>>(
      Xb, WT, bq, bk, bv, tab, Qr, Kr, Vr);

  attn<<<dim3(T_SEQ / QTILE, B_SZ * HKV), 256, 0, stream>>>(Qr, Kr, Vr, out);
}
